// CausalMultiHeadAttention_45561013076090
// MI455X (gfx1250) — compile-verified, matrix operand order checked by reading, not run on hardware
//
#include <hip/hip_runtime.h>
#include <hip/hip_bf16.h>

typedef __attribute__((ext_vector_type(16))) __bf16 v16bf;
typedef __attribute__((ext_vector_type(8)))  __bf16 v8bf;
typedef __attribute__((ext_vector_type(4)))  __bf16 v4bf;
typedef __attribute__((ext_vector_type(8)))  float  v8f;

#define B_   4
#define T_   2048
#define DM_  1024
#define H_   16
#define DK_  64
#define WIN_ 512

// ---------------- fragment helpers (wave32 WMMA layouts, ISA 7.12.2) ----------------

// A/B fragment load from a row-major matrix: lane L (<16) -> row r0+L, K = {k0..k0+7, k0+16..k0+23};
// lane L+16 -> row r0+L, K = {k0+8..k0+15, k0+24..k0+31}.  Two 16-byte loads per lane.
__device__ __forceinline__ v16bf frag_ld_row(const __bf16* base, int ld, int row, int k0, int lane) {
    int ko = k0 + ((lane >> 4) << 3);
    const __bf16* p = base + (size_t)row * ld + ko;
    v8bf lo = *(const v8bf*)(p);
    v8bf hi = *(const v8bf*)(p + 16);
    return __builtin_shufflevector(lo, hi, 0,1,2,3,4,5,6,7,8,9,10,11,12,13,14,15);
}

__device__ __forceinline__ v16bf frag_ld(const __bf16* base, int ld, int r0, int k0, int lane) {
    return frag_ld_row(base, ld, r0 + (lane & 15), k0, lane);
}

__device__ __forceinline__ v8f wmma_bf16(v16bf a, v16bf b, v8f c) {
    // 8 args: (neg_a, A, neg_b, B, c_mod, C, reuse_a, reuse_b)
    return __builtin_amdgcn_wmma_f32_16x16x32_bf16(false, a, false, b, (short)0, c, false, false);
}

// ---------------- f32 -> bf16 conversion ----------------

__global__ void cvt_f32_bf16(const float* __restrict__ s, __bf16* __restrict__ d, int n) {
    int i = (blockIdx.x * blockDim.x + threadIdx.x) * 4;
    if (i + 3 < n) {
        float4 v = *(const float4*)(s + i);
        v4bf o;
        o[0] = (__bf16)v.x; o[1] = (__bf16)v.y; o[2] = (__bf16)v.z; o[3] = (__bf16)v.w;
        *(v4bf*)(d + i) = o;
    }
}

// ---------------- GEMM: O[m,n] = sum_k A[m,k] * W[n,k]  (i.e. A @ W^T) ----------------
// One wave computes a 32x64 tile: 2 M-subtiles x 4 N-subtiles, 8 WMMAs per K=32 step.
// B-fragments amortized over both M-subtiles (~21 FLOP/byte from L2).
// MODE 0: epilogue RoPE, store bf16 [b,h,t,d]             (Q and K projections)
// MODE 1: epilogue plain, store bf16 transposed [b,h,d,t] (V projection)
// MODE 2: A is attention output in [b,h,t,d] layout (head-aware K addressing),
//         store f32 [m, n] to d_out                       (output projection)
template<int MODE>
__global__ void gemm16(const __bf16* __restrict__ A, const __bf16* __restrict__ Wb,
                       void* __restrict__ Out, const int* __restrict__ tp) {
    const int lane = threadIdx.x;
    const int n0 = blockIdx.x * 64;
    const int m0 = blockIdx.y * 32;              // 32 rows, never crosses a batch (T_%32==0)
    const int hi8 = (lane >> 4) << 3;

    v8f c[2][4] = {};

    for (int kk = 0; kk < DM_; kk += 32) {
        v16bf a0, a1;
        if (MODE == 2) {
            // A stored [b][h][t][d]: k = h*64 + d ; kk is 32-aligned so block stays in one head
            int b = m0 >> 11;
            int h = kk >> 6;
            int ko = (kk & 63) + ((lane >> 4) << 3);
            #pragma unroll
            for (int ms = 0; ms < 2; ++ms) {
                int t = ((m0 + ms * 16) & (T_ - 1)) + (lane & 15);
                const __bf16* p = A + ((size_t)(b * H_ + h) * T_ + t) * DK_ + ko;
                v8bf lo = *(const v8bf*)(p);
                v8bf hi = *(const v8bf*)(p + 16);
                v16bf a = __builtin_shufflevector(lo, hi, 0,1,2,3,4,5,6,7,8,9,10,11,12,13,14,15);
                if (ms == 0) a0 = a; else a1 = a;
            }
        } else {
            a0 = frag_ld(A, DM_, m0,      kk, lane);
            a1 = frag_ld(A, DM_, m0 + 16, kk, lane);
        }
        #pragma unroll
        for (int f = 0; f < 4; ++f) {
            v16bf bfr = frag_ld(Wb, DM_, n0 + f * 16, kk, lane);
            c[0][f] = wmma_bf16(a0, bfr, c[0][f]);
            c[1][f] = wmma_bf16(a1, bfr, c[1][f]);
        }
    }

    #pragma unroll
    for (int ms = 0; ms < 2; ++ms) {
        const int m0s = m0 + ms * 16;
        const int b = m0s >> 11;

        if (MODE == 0) {
            // RoPE + store bf16 [b,h,t,d].  C layout: lane holds column n, rows m = r + hi8.
            // Channel pair (2i, 2i+1) sits in adjacent lanes -> exchange via shfl_xor(.,1).
            __bf16* dst = (__bf16*)Out;
            #pragma unroll
            for (int f = 0; f < 4; ++f) {
                int n = n0 + f * 16 + (lane & 15);
                int h = n >> 6, d = n & 63;
                float invf = __expf(-(float)(d & ~1) * (9.210340372f / (float)DK_)); // theta^(-2i/dk)
                v8f cv = c[ms][f];
                #pragma unroll
                for (int r = 0; r < 8; ++r) {
                    int m = m0s + r + hi8;
                    int t = m & (T_ - 1);
                    float ang = (float)tp[t] * invf;
                    float sn = __sinf(ang), csv = __cosf(ang);
                    float own = cv[r];
                    float prt = __shfl_xor(own, 1, 32);
                    float val = ((lane & 1) == 0) ? (own * csv - prt * sn)   // x1*cos - x2*sin
                                                  : (prt * sn + own * csv);  // x1*sin + x2*cos
                    dst[((size_t)(b * H_ + h) * T_ + t) * DK_ + d] = (__bf16)val;
                }
            }
        } else if (MODE == 1) {
            // V stored transposed [b,h,d,t]; lane's 8 rows are contiguous t -> one 16B store.
            __bf16* dst = (__bf16*)Out;
            int tbase = (m0s & (T_ - 1)) + hi8;
            #pragma unroll
            for (int f = 0; f < 4; ++f) {
                int n = n0 + f * 16 + (lane & 15);
                int h = n >> 6, d = n & 63;
                v8f cv = c[ms][f];
                v8bf pk;
                #pragma unroll
                for (int r = 0; r < 8; ++r) pk[r] = (__bf16)cv[r];
                *(v8bf*)(dst + ((size_t)(b * H_ + h) * DK_ + d) * T_ + tbase) = pk;
            }
        } else {
            float* dst = (float*)Out;
            #pragma unroll
            for (int f = 0; f < 4; ++f) {
                int n = n0 + f * 16 + (lane & 15);
                v8f cv = c[ms][f];
                #pragma unroll
                for (int r = 0; r < 8; ++r) {
                    int m = m0s + r + hi8;
                    dst[(size_t)m * DM_ + n] = cv[r];
                }
            }
        }
    }
}

// ---------------- Flash-style windowed causal attention ----------------
// One wave per (b*h, 16-row q tile). 32 keys per iteration:
//   S(16x32) = Q K^T (4 WMMAs), mask+scale, online softmax, P -> LDS -> A-frag,
//   O(16x64) += P V (4 WMMAs with transposed-V B-fragments).
__global__ void attn_kernel(const __bf16* __restrict__ qb, const __bf16* __restrict__ kb,
                            const __bf16* __restrict__ vtb, __bf16* __restrict__ ob) {
    __shared__ __align__(16) __bf16 pl[16 * 32];
    const int lane = threadIdx.x;
    const int q0 = blockIdx.x * 16;
    const int bh = blockIdx.y;
    const int hi8 = (lane >> 4) << 3;
    const float NEGINF = -__builtin_inff();

    const __bf16* qh = qb  + (size_t)bh * T_ * DK_;   // [t][d]
    const __bf16* kh = kb  + (size_t)bh * T_ * DK_;   // [t][d]
    const __bf16* vh = vtb + (size_t)bh * DK_ * T_;   // [d][t]

    v16bf aq0 = frag_ld(qh, DK_, q0, 0,  lane);
    v16bf aq1 = frag_ld(qh, DK_, q0, 32, lane);

    v8f o0 = {}, o1 = {}, o2 = {}, o3 = {};
    float mrow[8], lrow[8];
    #pragma unroll
    for (int r = 0; r < 8; ++r) { mrow[r] = NEGINF; lrow[r] = 0.f; }

    int jstart = q0 - WIN_; if (jstart < 0) jstart = 0;
    const float scale = 0.125f;   // 1/sqrt(64)

    for (int j0 = jstart; j0 < q0 + 16; j0 += 32) {
        // ---- scores: two 16-key subtiles, K reduced over d=64 (2 chained WMMAs each) ----
        v8f s0 = {}, s1 = {};
        {
            int row = j0 + (lane & 15);            if (row > T_ - 1) row = T_ - 1;
            v16bf bk0 = frag_ld_row(kh, DK_, row, 0,  lane);
            v16bf bk1 = frag_ld_row(kh, DK_, row, 32, lane);
            s0 = wmma_bf16(aq0, bk0, s0);
            s0 = wmma_bf16(aq1, bk1, s0);
        }
        {
            int row = j0 + 16 + (lane & 15);       if (row > T_ - 1) row = T_ - 1;
            v16bf bk0 = frag_ld_row(kh, DK_, row, 0,  lane);
            v16bf bk1 = frag_ld_row(kh, DK_, row, 32, lane);
            s1 = wmma_bf16(aq0, bk0, s1);
            s1 = wmma_bf16(aq1, bk1, s1);
        }

        // ---- mask (causal + window) and scale ----
        int jA = j0 + (lane & 15);
        int jB = j0 + 16 + (lane & 15);
        #pragma unroll
        for (int r = 0; r < 8; ++r) {
            int i = q0 + r + hi8;
            s0[r] = ((i >= jA) && (i - jA <= WIN_)) ? s0[r] * scale : NEGINF;
            s1[r] = ((i >= jB) && (i - jB <= WIN_)) ? s1[r] * scale : NEGINF;
        }

        // ---- online softmax update.  Row reductions stay within a 16-lane half; the
        // m==-inf branch is uniform per half (row stats identical across its 16 lanes),
        // and every shfl source lane lies in the same half, so divergence is safe. ----
        #pragma unroll
        for (int r = 0; r < 8; ++r) {
            float v = fmaxf(s0[r], s1[r]);
            v = fmaxf(v, __shfl_xor(v, 1, 32));
            v = fmaxf(v, __shfl_xor(v, 2, 32));
            v = fmaxf(v, __shfl_xor(v, 4, 32));
            v = fmaxf(v, __shfl_xor(v, 8, 32));
            float mn = fmaxf(mrow[r], v);
            if (mn == NEGINF) {                 // no allowed keys yet for this row
                s0[r] = 0.f; s1[r] = 0.f;
                continue;
            }
            float alpha = __expf(mrow[r] - mn);  // exp(-inf)=0 on first contribution
            mrow[r] = mn;
            float p0 = __expf(s0[r] - mn);
            float p1 = __expf(s1[r] - mn);
            float rs = p0 + p1;
            rs += __shfl_xor(rs, 1, 32); rs += __shfl_xor(rs, 2, 32);
            rs += __shfl_xor(rs, 4, 32); rs += __shfl_xor(rs, 8, 32);
            lrow[r] = lrow[r] * alpha + rs;
            o0[r] *= alpha; o1[r] *= alpha; o2[r] *= alpha; o3[r] *= alpha;
            s0[r] = p0; s1[r] = p1;
        }

        // ---- P (C layout) -> LDS -> A fragment (lane = q row) ----
        #pragma unroll
        for (int r = 0; r < 8; ++r) {
            int mr = r + hi8;
            pl[mr * 32 + (lane & 15)]      = (__bf16)s0[r];
            pl[mr * 32 + 16 + (lane & 15)] = (__bf16)s1[r];
        }
        __syncthreads();
        v16bf ap;
        {
            int ko = (lane >> 4) << 3;
            const __bf16* p = &pl[(lane & 15) * 32 + ko];
            v8bf lo = *(const v8bf*)(p);
            v8bf hi = *(const v8bf*)(p + 16);
            ap = __builtin_shufflevector(lo, hi, 0,1,2,3,4,5,6,7,8,9,10,11,12,13,14,15);
        }
        __syncthreads();

        // ---- O += P @ V : B-fragments from transposed V (contiguous 16B loads) ----
        int ko = (lane >> 4) << 3;
        int t0a = j0 + ko;      if (t0a > T_ - 8) t0a = T_ - 8;   // clamped; P==0 there
        int t0b = j0 + ko + 16; if (t0b > T_ - 8) t0b = T_ - 8;
        #pragma unroll
        for (int f = 0; f < 4; ++f) {
            const __bf16* p = vh + (size_t)(f * 16 + (lane & 15)) * T_;
            v8bf lo = *(const v8bf*)(p + t0a);
            v8bf hi = *(const v8bf*)(p + t0b);
            v16bf bv = __builtin_shufflevector(lo, hi, 0,1,2,3,4,5,6,7,8,9,10,11,12,13,14,15);
            if (f == 0) o0 = wmma_bf16(ap, bv, o0);
            if (f == 1) o1 = wmma_bf16(ap, bv, o1);
            if (f == 2) o2 = wmma_bf16(ap, bv, o2);
            if (f == 3) o3 = wmma_bf16(ap, bv, o3);
        }
    }

    // ---- normalize and store bf16 [b,h,t,d] ----
    v8f* os[4] = { &o0, &o1, &o2, &o3 };
    #pragma unroll
    for (int f = 0; f < 4; ++f) {
        int d = f * 16 + (lane & 15);
        #pragma unroll
        for (int r = 0; r < 8; ++r) {
            int t = q0 + r + hi8;
            ob[((size_t)bh * T_ + t) * DK_ + d] = (__bf16)((*os[f])[r] / lrow[r]);
        }
    }
}

// ---------------- host launch ----------------
extern "C" void kernel_launch(void* const* d_in, const int* in_sizes, int n_in,
                              void* d_out, int out_size, void* d_ws, size_t ws_size,
                              hipStream_t stream) {
    const float* x  = (const float*)d_in[0];
    const int*   tp = (const int*)d_in[1];
    const float* Wq = (const float*)d_in[2];
    const float* Wk = (const float*)d_in[3];
    const float* Wv = (const float*)d_in[4];
    const float* Wo = (const float*)d_in[5];

    // workspace layout (bf16), ~88 MB total
    char* ws = (char*)d_ws;
    __bf16* xb   = (__bf16*)ws;                                   // 8192*1024
    __bf16* wqb  = (__bf16*)(ws + (size_t)16 * 1024 * 1024);      // 1024*1024 each
    __bf16* wkb  = wqb + 1024 * 1024;
    __bf16* wvb  = wkb + 1024 * 1024;
    __bf16* wob  = wvb + 1024 * 1024;
    __bf16* qb   = (__bf16*)(ws + (size_t)24 * 1024 * 1024);      // [b,h,t,d]
    __bf16* kbp  = qb  + (size_t)B_ * H_ * T_ * DK_;              // [b,h,t,d]
    __bf16* vtb  = kbp + (size_t)B_ * H_ * T_ * DK_;              // [b,h,d,t]
    __bf16* attb = vtb + (size_t)B_ * H_ * T_ * DK_;              // [b,h,t,d]

    // 1) convert to bf16
    cvt_f32_bf16<<<dim3((B_ * T_ * DM_) / 1024), dim3(256), 0, stream>>>(x,  xb,  B_ * T_ * DM_);
    cvt_f32_bf16<<<dim3((DM_ * DM_) / 1024),     dim3(256), 0, stream>>>(Wq, wqb, DM_ * DM_);
    cvt_f32_bf16<<<dim3((DM_ * DM_) / 1024),     dim3(256), 0, stream>>>(Wk, wkb, DM_ * DM_);
    cvt_f32_bf16<<<dim3((DM_ * DM_) / 1024),     dim3(256), 0, stream>>>(Wv, wvb, DM_ * DM_);
    cvt_f32_bf16<<<dim3((DM_ * DM_) / 1024),     dim3(256), 0, stream>>>(Wo, wob, DM_ * DM_);

    // 2) QKV projections (RoPE fused into Q/K epilogue, V stored transposed)
    dim3 ggrid(DM_ / 64, (B_ * T_) / 32), gblk(32);
    gemm16<0><<<ggrid, gblk, 0, stream>>>(xb, wqb, qb,  tp);
    gemm16<0><<<ggrid, gblk, 0, stream>>>(xb, wkb, kbp, tp);
    gemm16<1><<<ggrid, gblk, 0, stream>>>(xb, wvb, vtb, tp);

    // 3) windowed causal attention
    attn_kernel<<<dim3(T_ / 16, B_ * H_), gblk, 0, stream>>>(qb, kbp, vtb, attb);

    // 4) output projection -> f32 d_out
    gemm16<2><<<ggrid, gblk, 0, stream>>>(attb, wob, d_out, tp);
}
